// ImprovedPairwiseAttention_39041252720810
// MI455X (gfx1250) — matrix operand order repaired and checked by reading, not compiled, not run on hardware
//
#include <hip/hip_runtime.h>
#include <math.h>

// Problem constants (match reference)
#define DD    1024
#define HH    16
#define HDIM  64
#define BB    8
#define NN    1024
#define NTOK  (BB * NN)   // 8192
#define LN_EPS 1e-5f

typedef __attribute__((ext_vector_type(16))) _Float16 v16h;
typedef __attribute__((ext_vector_type(8)))  _Float16 v8h;
typedef __attribute__((ext_vector_type(8)))  float    v8f;

__device__ __forceinline__ v8f wmma_f16(v16h a, v16h b, v8f c) {
  // D = A(16x32,f16) * B(32x16,f16) + C(16x16,f32)
  return __builtin_amdgcn_wmma_f32_16x16x32_f16(
      /*neg_a=*/false, a, /*neg_b=*/false, b,
      /*c_mod=*/(short)0, c, /*reuse_a=*/false, /*reuse_b=*/false);
}

// A-tile load, row-major f16 source. lane row = lane&15, kba = (lane>>4)*8.
// Elements 0..15 <- K = k0+2*kba..+15 (contiguous, matches B)
__device__ __forceinline__ v16h load_a(const _Float16* rowbase, int k0, int kba) {
  v8h lo = *(const v8h*)(rowbase + k0 + 2 * kba);
  v8h hi = *(const v8h*)(rowbase + k0 + 2 * kba + 8);
  v16h a;
#pragma unroll
  for (int j = 0; j < 8; ++j) { a[j] = lo[j]; a[j + 8] = hi[j]; }
  return a;
}

// ---------------------------------------------------------------------------
// Kernel 0: fp32 -> f16 conversions; transpose weights so B-tiles are
// contiguous per-lane (WT[n][k] = W[k][n]).
// ---------------------------------------------------------------------------
__global__ void prep_kernel(const float* __restrict__ x,
                            const float* __restrict__ Wq, const float* __restrict__ Wk,
                            const float* __restrict__ Wv, const float* __restrict__ Wo,
                            _Float16* __restrict__ Xh,
                            _Float16* __restrict__ WqT, _Float16* __restrict__ WkT,
                            _Float16* __restrict__ WvT, _Float16* __restrict__ WoT) {
  long tid = (long)blockIdx.x * blockDim.x + threadIdx.x;
  long stride = (long)gridDim.x * blockDim.x;
  for (long i = tid; i < (long)NTOK * DD; i += stride) Xh[i] = (_Float16)x[i];
  for (long i = tid; i < (long)DD * DD; i += stride) {
    int k = (int)(i / DD), n = (int)(i % DD);
    long t = (long)n * DD + k;
    WqT[t] = (_Float16)Wq[i];
    WkT[t] = (_Float16)Wk[i];
    WvT[t] = (_Float16)Wv[i];
    WoT[t] = (_Float16)Wo[i];
  }
}

// ---------------------------------------------------------------------------
// Kernel 1: fused QKV projection GEMM. X[8192,1024] x {Wq|Wk|Wv}T -> f16.
// Q,K stored [b,h,n,hd] (row-major hd => natural A and B=K^T tiles).
// V stored transposed [b,h,hd,n] so PV B-tiles are contiguous per-lane.
// Grid: 512 M-tiles * 48 n-blocks(64 wide); 4 waves, 1 16x16 tile each.
// ---------------------------------------------------------------------------
__global__ __launch_bounds__(128)
void qkv_kernel(const _Float16* __restrict__ Xh,
                const _Float16* __restrict__ WqT, const _Float16* __restrict__ WkT,
                const _Float16* __restrict__ WvT,
                const float* __restrict__ bq, const float* __restrict__ bk,
                const float* __restrict__ bv,
                _Float16* __restrict__ Qb, _Float16* __restrict__ Kb,
                _Float16* __restrict__ Vt) {
  const int mt = blockIdx.x / 48;          // 0..511  (16 tokens each)
  const int nb = blockIdx.x % 48;          // 64-wide n block across 3*1024
  const int wave = threadIdx.x >> 5;
  const int lane = threadIdx.x & 31;
  const int nglob = nb * 64 + wave * 16;   // [0, 3072)
  const int mat = nglob / DD;              // 0=Q 1=K 2=V
  const int n0 = nglob % DD;

  const _Float16* WT = (mat == 0) ? WqT : (mat == 1) ? WkT : WvT;
  const float* bias  = (mat == 0) ? bq  : (mat == 1) ? bk  : bv;

  const int row = lane & 15;
  const int kba = (lane >> 4) * 8;
  const int col = lane & 15;
  const int kbb = (lane >> 4) * 16;

  const _Float16* Abase = Xh + (long)(mt * 16 + row) * DD;
  const _Float16* Bbase = WT + (long)(n0 + col) * DD;

  v8f c = {};
  for (int k0 = 0; k0 < DD; k0 += 32) {
    v16h a = load_a(Abase, k0, kba);
    v16h b = *(const v16h*)(Bbase + k0 + kbb);
    c = wmma_f16(a, b, c);
  }

  const int nc = n0 + col;
  const float bsv = bias[nc];
  const int hh = nc / HDIM, dd = nc % HDIM;
#pragma unroll
  for (int e = 0; e < 8; ++e) {
    const int m = e + ((lane >> 4) << 3);          // C layout: M = e (+8 hi half)
    const int tok = mt * 16 + m;
    const int bidx = tok / NN, nt = tok % NN;
    const _Float16 val = (_Float16)(c[e] + bsv);
    const long bh = (long)bidx * HH + hh;
    if (mat == 0)      Qb[(bh * NN + nt) * HDIM + dd] = val;
    else if (mat == 1) Kb[(bh * NN + nt) * HDIM + dd] = val;
    else               Vt[(bh * HDIM + dd) * NN + nt] = val;
  }
}

// ---------------------------------------------------------------------------
// Kernel 2: attention. One WG (4 waves) per (b,h, 16-query tile).
// scores (WMMA, +pair_bias, *1/8) -> LDS fp32 [16x1024] -> softmax (fp32,
// shuffle reductions) -> write attn (coalesced float4) -> PV (WMMA).
// ---------------------------------------------------------------------------
__global__ __launch_bounds__(128)
void attn_kernel(const _Float16* __restrict__ Qb, const _Float16* __restrict__ Kb,
                 const _Float16* __restrict__ Vt, const float* __restrict__ pair_bias,
                 float* __restrict__ attn, _Float16* __restrict__ AO) {
  __shared__ float S[16][NN];   // 64 KB (LDS budget is 320 KB/WGP)

  const int bh = blockIdx.x / (NN / 16);   // 0..127 = b*H+h
  const int qt = blockIdx.x % (NN / 16);   // 0..63
  const int q0 = qt * 16;
  const int wave = threadIdx.x >> 5;
  const int lane = threadIdx.x & 31;
  const int row = lane & 15;
  const int kba = (lane >> 4) * 8;
  const int col = lane & 15;
  const int kbb = (lane >> 4) * 16;

  // Q tile (16 x 64) = two A tiles, resident for the whole stripe
  const _Float16* Qrow = Qb + ((long)bh * NN + q0 + row) * HDIM;
  const v16h a0 = load_a(Qrow, 0, kba);
  const v16h a1 = load_a(Qrow, 32, kba);

  // -------- scores: each wave covers key tiles wave, wave+4, ... ----------
  const float scale = 0.125f;  // 1/sqrt(64)
  for (int it = wave; it < NN / 16; it += 4) {
    const int kk0 = it * 16;
    // B = K^T comes for free from K's [token][hd] layout
    const _Float16* Krow = Kb + ((long)bh * NN + kk0 + col) * HDIM;
    v16h b0 = *(const v16h*)(Krow + kbb);
    v16h b1 = *(const v16h*)(Krow + 32 + kbb);
    v8f c = {};
    c = wmma_f16(a0, b0, c);
    c = wmma_f16(a1, b1, c);
#pragma unroll
    for (int e = 0; e < 8; ++e) {
      const int m = e + ((lane >> 4) << 3);
      S[m][kk0 + col] = c[e] * scale + pair_bias[(long)(q0 + m) * DD + kk0 + col];
    }
  }
  __syncthreads();

  // -------- softmax: 8 lanes per row (lane-shuffle reductions) ------------
  {
    const int r = threadIdx.x >> 3;
    const int c0 = (threadIdx.x & 7) * 128;
    float* Srow = &S[r][0];
    float mx = -3.0e38f;
    for (int i = 0; i < 128; ++i) mx = fmaxf(mx, Srow[c0 + i]);
#pragma unroll
    for (int off = 1; off < 8; off <<= 1) mx = fmaxf(mx, __shfl_xor(mx, off, 8));
    float sum = 0.f;
    for (int i = 0; i < 128; ++i) {
      float e = __expf(Srow[c0 + i] - mx);
      Srow[c0 + i] = e;
      sum += e;
    }
#pragma unroll
    for (int off = 1; off < 8; off <<= 1) sum += __shfl_xor(sum, off, 8);
    const float inv = 1.0f / sum;
    float* arow = attn + ((long)bh * NN + q0 + r) * NN + c0;
    for (int i = 0; i < 128; i += 4) {           // normalize + write attn out
      float4 p = *(float4*)&Srow[c0 + i];
      p.x *= inv; p.y *= inv; p.z *= inv; p.w *= inv;
      *(float4*)&Srow[c0 + i] = p;
      *(float4*)&arow[i] = p;
    }
  }
  __syncthreads();

  // -------- out = P @ V : wave w computes hd columns [16w, 16w+16) --------
  v8f o = {};
  const _Float16* Vbase = Vt + ((long)bh * HDIM + wave * 16 + col) * NN;
  for (int k0 = 0; k0 < NN; k0 += 32) {
    v16h pa;
#pragma unroll
    for (int j = 0; j < 8; ++j) {
      pa[j]     = (_Float16)S[row][k0 + 2 * kba + j];
      pa[j + 8] = (_Float16)S[row][k0 + 2 * kba + 8 + j];
    }
    v16h vb = *(const v16h*)(Vbase + k0 + kbb);
    o = wmma_f16(pa, vb, o);
  }
  const int h = bh % HH, b_ = bh / HH;
#pragma unroll
  for (int e = 0; e < 8; ++e) {
    const int m = e + ((lane >> 4) << 3);
    const long tok = (long)b_ * NN + q0 + m;
    AO[tok * DD + h * HDIM + wave * 16 + col] = (_Float16)o[e];
  }
}

// ---------------------------------------------------------------------------
// Kernel 3: output projection (WMMA) + bias + residual + LayerNorm.
// One WG per 16-token tile; 4 waves sweep the 64 n-tiles; row stats via
// shuffle reductions over the fp32 row held in LDS.
// ---------------------------------------------------------------------------
__global__ __launch_bounds__(128)
void oproj_ln_kernel(const _Float16* __restrict__ AO, const _Float16* __restrict__ WoT,
                     const float* __restrict__ bo, const float* __restrict__ x,
                     const float* __restrict__ gamma, const float* __restrict__ beta,
                     float* __restrict__ out) {
  __shared__ float Y[16][DD];   // 64 KB

  const int mt = blockIdx.x;               // 0..511
  const int wave = threadIdx.x >> 5;
  const int lane = threadIdx.x & 31;
  const int row = lane & 15;
  const int kba = (lane >> 4) * 8;
  const int col = lane & 15;
  const int kbb = (lane >> 4) * 16;

  const _Float16* Abase = AO + (long)(mt * 16 + row) * DD;

  for (int nt = wave; nt < 64; nt += 4) {
    const int n0 = nt * 16;
    const _Float16* Bbase = WoT + (long)(n0 + col) * DD;
    v8f c = {};
    for (int k0 = 0; k0 < DD; k0 += 32) {
      v16h a = load_a(Abase, k0, kba);
      v16h b = *(const v16h*)(Bbase + k0 + kbb);
      c = wmma_f16(a, b, c);
    }
    const float bv = bo[n0 + col];
#pragma unroll
    for (int e = 0; e < 8; ++e) {
      const int m = e + ((lane >> 4) << 3);
      Y[m][n0 + col] = c[e] + bv + x[(long)(mt * 16 + m) * DD + n0 + col];
    }
  }
  __syncthreads();

  // LayerNorm: 8 lanes per row
  const int r = threadIdx.x >> 3;
  const int c0 = (threadIdx.x & 7) * 128;
  float* Yrow = &Y[r][0];
  float s = 0.f, s2 = 0.f;
  for (int i = 0; i < 128; ++i) {
    const float v = Yrow[c0 + i];
    s += v; s2 += v * v;
  }
#pragma unroll
  for (int off = 1; off < 8; off <<= 1) {
    s  += __shfl_xor(s,  off, 8);
    s2 += __shfl_xor(s2, off, 8);
  }
  const float mu   = s * (1.0f / 1024.0f);
  const float var  = s2 * (1.0f / 1024.0f) - mu * mu;
  const float rstd = rsqrtf(var + LN_EPS);
  float* orow = out + (long)(mt * 16 + r) * DD + c0;
  for (int i = 0; i < 128; ++i)
    orow[i] = (Yrow[c0 + i] - mu) * rstd * gamma[c0 + i] + beta[c0 + i];
}

// ---------------------------------------------------------------------------
extern "C" void kernel_launch(void* const* d_in, const int* in_sizes, int n_in,
                              void* d_out, int out_size, void* d_ws, size_t ws_size,
                              hipStream_t stream) {
  const float* x     = (const float*)d_in[0];
  const float* Wq    = (const float*)d_in[1];
  const float* bq    = (const float*)d_in[2];
  const float* Wk    = (const float*)d_in[3];
  const float* bk    = (const float*)d_in[4];
  const float* Wv    = (const float*)d_in[5];
  const float* bv    = (const float*)d_in[6];
  const float* pb    = (const float*)d_in[7];
  const float* Wo    = (const float*)d_in[8];
  const float* bo    = (const float*)d_in[9];
  const float* gamma = (const float*)d_in[10];
  const float* beta  = (const float*)d_in[11];

  float* out  = (float*)d_out;                       // [8,1024,1024]
  float* attn = out + (size_t)BB * NN * DD;          // [8,16,1024,1024]

  // Workspace carve (~88 MB total, all 16B-aligned slices)
  char* ws = (char*)d_ws;
  _Float16* Xh  = (_Float16*)ws; ws += (size_t)NTOK * DD * 2;
  _Float16* WqT = (_Float16*)ws; ws += (size_t)DD * DD * 2;
  _Float16* WkT = (_Float16*)ws; ws += (size_t)DD * DD * 2;
  _Float16* WvT = (_Float16*)ws; ws += (size_t)DD * DD * 2;
  _Float16* WoT = (_Float16*)ws; ws += (size_t)DD * DD * 2;
  _Float16* Qb  = (_Float16*)ws; ws += (size_t)NTOK * DD * 2;  // [b,h,n,hd]
  _Float16* Kb  = (_Float16*)ws; ws += (size_t)NTOK * DD * 2;  // [b,h,n,hd]
  _Float16* Vt  = (_Float16*)ws; ws += (size_t)NTOK * DD * 2;  // [b,h,hd,n]
  _Float16* AO  = (_Float16*)ws; ws += (size_t)NTOK * DD * 2;  // [tok,feat]

  prep_kernel<<<2048, 256, 0, stream>>>(x, Wq, Wk, Wv, Wo, Xh, WqT, WkT, WvT, WoT);
  qkv_kernel<<<(NTOK / 16) * 48, 128, 0, stream>>>(Xh, WqT, WkT, WvT, bq, bk, bv,
                                                   Qb, Kb, Vt);
  attn_kernel<<<BB * HH * (NN / 16), 128, 0, stream>>>(Qb, Kb, Vt, pb, attn, AO);
  oproj_ln_kernel<<<NTOK / 16, 128, 0, stream>>>(AO, WoT, bo, x, gamma, beta, out);
}
